// FourierSelfAtt_67637144978382
// MI455X (gfx1250) — matrix operand order repaired and checked by reading, not compiled, not run on hardware
//
#include <hip/hip_runtime.h>

// FourierSelfAtt for MI455X (gfx1250): modes-only DFT as f16 WMMA GEMMs,
// LDS-tiled so q/k/v stream from HBM exactly once.
// B=8,H=8 -> BH=64 ; L=4096 ; D=64 ; MODES=64 ; SCALE = 1/8.

typedef __attribute__((ext_vector_type(16))) _Float16 v16h;
typedef __attribute__((ext_vector_type(8)))  float    v8f;

#define LSEQ   4096
#define NMODE  64
#define DHEAD  64
#define NBH    64
#define FSCALE 0.125f
#define TWO_PI_OVER_L (6.283185307179586f / 4096.0f)

union HF16 { v16h v; unsigned int u[8]; };

static __device__ __forceinline__ v8f wmma16(v16h a, v16h b, v8f c) {
  return __builtin_amdgcn_wmma_f32_16x16x32_f16(false, a, false, b, (short)0, c,
                                                false, false);
}

static __device__ __forceinline__ unsigned int pack_h2(float a, float b) {
  union { _Float16 h[2]; unsigned int u; } p;
  p.h[0] = (_Float16)a;
  p.h[1] = (_Float16)b;
  return p.u;
}

// ---- generic per-element fragment gathers (used by the tiny attention GEMMs)
// A (16x32 f16): lane r=lane&15 row M=r; lanes0-15 V0..3->K0..7,V4..7->K16..23;
// lanes16-31 +8.  B (32x16): lane c col N; same K order as A.
template <typename T>
static __device__ __forceinline__ v16h
load_a_frag(const T* p, int srow, int row0, int k0, float mul) {
  int lane = threadIdx.x & 31;
  int r = lane & 15, hi = lane >> 4;
  const T* base = p + (long)(row0 + r) * srow + k0;
  v16h a;
#pragma unroll
  for (int v = 0; v < 8; ++v) {
#pragma unroll
    for (int h = 0; h < 2; ++h) {
      int k = ((v < 4) ? 0 : 16) + hi * 8 + 2 * (v & 3) + h;
      a[2 * v + h] = (_Float16)(mul * (float)base[k]);
    }
  }
  return a;
}

template <typename T>
static __device__ __forceinline__ v16h
load_b_frag(const T* p, int sk, int scol, int k0, int col0) {
  int lane = threadIdx.x & 31;
  int c = lane & 15, hi = lane >> 4;
  const T* base = p + (long)k0 * sk + (long)(col0 + c) * scol;
  v16h b;
#pragma unroll
  for (int v = 0; v < 8; ++v) {
#pragma unroll
    for (int h = 0; h < 2; ++h) {
      int k = ((v < 4) ? 0 : 16) + hi * 8 + 2 * (v & 3) + h;
      b[2 * v + h] = (_Float16)(float)base[(long)k * sk];
    }
  }
  return b;
}

// LDS fragment builds: k-pairs are contiguous f16 -> 8 u32 ds loads each.
static __device__ __forceinline__ v16h lds_a_frag(const _Float16* rowbase, int hi) {
  HF16 a;
#pragma unroll
  for (int v = 0; v < 8; ++v) {
    int keven = ((v < 4) ? 0 : 16) + hi * 8 + 2 * (v & 3);
    a.u[v] = *(const unsigned int*)&rowbase[keven];
  }
  return a.v;
}
static __device__ __forceinline__ v16h lds_b_frag(const _Float16* colbase, int hi) {
  HF16 b;
#pragma unroll
  for (int v = 0; v < 8; ++v) {
    int keven = ((v < 4) ? 0 : 16) + hi * 8 + 2 * (v & 3);
    b.u[v] = *(const unsigned int*)&colbase[keven];
  }
  return b.v;
}

// C/D (16x16 f32): lane c col; VGPR v -> row v + 8*(lane>>4).
static __device__ __forceinline__ void
store_c_frag(float* p, int srow, int row0, int col0, v8f c) {
  int lane = threadIdx.x & 31;
  int cc = lane & 15, hi = lane >> 4;
#pragma unroll
  for (int v = 0; v < 8; ++v)
    p[(long)(row0 + v + 8 * hi) * srow + (col0 + cc)] = c[v];
}

// ---- complex tanh: (sinh 2x + i sin 2y) / (cosh 2x + cos 2y) ---------------
static __device__ __forceinline__ void ctanh_f(float x, float y, float& rr, float& ri) {
  float tx = 2.0f * x, ty = 2.0f * y;
  if (fabsf(tx) > 40.0f) { rr = (tx > 0.0f) ? 1.0f : -1.0f; ri = 0.0f; return; }
  float sh = sinhf(tx), ch = coshf(tx);
  float sn = __sinf(ty), cs = __cosf(ty);
  float inv = 1.0f / (ch + cs);
  rr = sh * inv;
  ri = sn * inv;
}

// ---- K1a: forward DFT weights per selected mode  W[tensor][re/im][m][t] ----
__global__ void build_fw_weights(const int* __restrict__ idx_q,
                                 const int* __restrict__ idx_k,
                                 const int* __restrict__ idx_v,
                                 _Float16* __restrict__ W) {
  int tid = blockIdx.x * blockDim.x + threadIdx.x;
  if (tid >= 3 * NMODE * LSEQ) return;
  int t = tid & (LSEQ - 1);
  int m = (tid >> 12) & (NMODE - 1);
  int tensor = tid >> 18;
  const int* idx = (tensor == 0) ? idx_q : ((tensor == 1) ? idx_k : idx_v);
  int mode = idx[m];
  int ph = (mode * t) & (LSEQ - 1);            // exact argument reduction
  float ang = (float)ph * TWO_PI_OVER_L;
  float s, c;
  __sincosf(ang, &s, &c);
  long base = (long)tensor * 2 * NMODE * LSEQ;
  W[base + (long)m * LSEQ + t] = (_Float16)c;
  W[base + (long)NMODE * LSEQ + (long)m * LSEQ + t] = (_Float16)(-s);
}

// ---- K1b: irfft weights  A2[t][0..63]=cos, A2[t][64..127]=-sin -------------
__global__ void build_inv_weights(_Float16* __restrict__ A2) {
  int tid = blockIdx.x * blockDim.x + threadIdx.x;
  if (tid >= LSEQ * NMODE) return;
  int m = tid & (NMODE - 1);
  int t = tid >> 6;
  int ph = (m * t) & (LSEQ - 1);
  float ang = (float)ph * TWO_PI_OVER_L;
  float s, c;
  __sincosf(ang, &s, &c);
  A2[(long)t * 128 + m] = (_Float16)c;
  A2[(long)t * 128 + 64 + m] = (_Float16)(-s);
}

// ---- K2: mode-DFT GEMM, LDS tiled. One block per (bh,tensor) computes the
//      full [2 part][64 m][64 d] output; x streamed from HBM exactly once. ---
#define WPAD 36   // f16 row stride (72B = 18 words, conflict-free, 4B-aligned)
__global__ __launch_bounds__(256) void
dft_gemm(const float* __restrict__ q, const float* __restrict__ k,
         const float* __restrict__ v, const _Float16* __restrict__ W,
         float* __restrict__ ft) {
  __shared__ _Float16 Ws[2 * NMODE * WPAD];  // [part][m][k 0..31]
  __shared__ _Float16 Xs[DHEAD * WPAD];      // [d][k 0..31] (transposed slab)
  int tid = threadIdx.x;
  int bid = blockIdx.x;                      // 0..191
  int tensor = bid % 3, bh = bid / 3;
  const float* xg = ((tensor == 0) ? q : (tensor == 1) ? k : v) +
                    (long)bh * LSEQ * DHEAD;
  const unsigned int* Wg =
      (const unsigned int*)(W + (long)tensor * 2 * NMODE * LSEQ);
  int w = tid >> 5, lane = tid & 31;
  int part = w >> 2, tm = w & 3;
  int r = lane & 15, hi = lane >> 4;
  v8f acc[4] = {};
  for (int k0 = 0; k0 < LSEQ; k0 += 32) {
    // stage W slab: 2*64*32 f16 = 2048 u32, coalesced
#pragma unroll
    for (int j = 0; j < 8; ++j) {
      int c = tid + j * 256;
      int kp = c & 15, m = (c >> 4) & 63, p = c >> 10;
      unsigned int d32 = Wg[(long)p * (NMODE * 2048) + m * 2048 + (k0 >> 1) + kp];
      *(unsigned int*)&Ws[p * (NMODE * WPAD) + m * WPAD + 2 * kp] = d32;
    }
    // stage x slab transposed + f32->f16: 32*64 elems = 1024 u32
#pragma unroll
    for (int j = 0; j < 4; ++j) {
      int c = tid + j * 256;
      int d = c & 63, kp = c >> 6;
      float x0 = xg[(long)(k0 + 2 * kp) * DHEAD + d];
      float x1 = xg[(long)(k0 + 2 * kp + 1) * DHEAD + d];
      *(unsigned int*)&Xs[d * WPAD + 2 * kp] = pack_h2(x0, x1);
      if (k0 + 32 < LSEQ)   // gfx1250 global_prefetch of the next slab
        __builtin_prefetch(&xg[(long)(k0 + 32 + 2 * kp) * DHEAD + d], 0, 0);
    }
    __syncthreads();
    v16h a = lds_a_frag(&Ws[part * (NMODE * WPAD) + (tm * 16 + r) * WPAD], hi);
#pragma unroll
    for (int tn = 0; tn < 4; ++tn) {
      v16h b = lds_b_frag(&Xs[(tn * 16 + r) * WPAD], hi);
      acc[tn] = wmma16(a, b, acc[tn]);
    }
    __syncthreads();
  }
  float* out = ft + ((long)tensor * NBH + bh) * 2 * NMODE * DHEAD +
               (long)part * NMODE * DHEAD;
#pragma unroll
  for (int tn = 0; tn < 4; ++tn)
    store_c_frag(out, DHEAD, tm * 16, tn * 16, acc[tn]);
}

// ---- K3: scores = ctanh( SCALE * q_ft @ k_ft^T )  (complex, tiny) ----------
__global__ void qk_tanh(const float* __restrict__ ft, float* __restrict__ scores) {
  int wid = blockIdx.x * (blockDim.x >> 5) + (threadIdx.x >> 5);
  if (wid >= NBH * 16) return;
  int tn = wid & 3, tm = (wid >> 2) & 3, bh = wid >> 4;
  const float* qr = ft + (long)bh * 2 * NMODE * DHEAD;
  const float* qi = qr + NMODE * DHEAD;
  const float* kr = ft + ((long)NBH + bh) * 2 * NMODE * DHEAD;
  const float* ki = kr + NMODE * DHEAD;
  v8f cre = {}, cim = {};
  for (int k0 = 0; k0 < DHEAD; k0 += 32) {
    v16h aqr  = load_a_frag(qr, DHEAD, tm * 16, k0, 1.0f);
    v16h aqi  = load_a_frag(qi, DHEAD, tm * 16, k0, 1.0f);
    v16h aqin = load_a_frag(qi, DHEAD, tm * 16, k0, -1.0f);
    v16h bkr  = load_b_frag(kr, 1, DHEAD, k0, tn * 16);  // k_ft[n][d] transposed
    v16h bki  = load_b_frag(ki, 1, DHEAD, k0, tn * 16);
    cre = wmma16(aqr, bkr, cre);
    cre = wmma16(aqin, bki, cre);   // qr*kr - qi*ki
    cim = wmma16(aqr, bki, cim);
    cim = wmma16(aqi, bkr, cim);    // qr*ki + qi*kr
  }
#pragma unroll
  for (int vv = 0; vv < 8; ++vv) {
    float rr, ri;
    ctanh_f(cre[vv] * FSCALE, cim[vv] * FSCALE, rr, ri);
    cre[vv] = rr;
    cim[vv] = ri;
  }
  float* s = scores + (long)bh * 2 * NMODE * NMODE;
  store_c_frag(s, NMODE, tm * 16, tn * 16, cre);
  store_c_frag(s + NMODE * NMODE, NMODE, tm * 16, tn * 16, cim);
}

// ---- K4: X2 = w_m * (scores @ v_ft) ; rows 0..63 Re, 64..127 Im ------------
__global__ void attn_v(const float* __restrict__ scores, const float* __restrict__ ft,
                       float* __restrict__ X2) {
  int wid = blockIdx.x * (blockDim.x >> 5) + (threadIdx.x >> 5);
  if (wid >= NBH * 16) return;
  int tn = wid & 3, tm = (wid >> 2) & 3, bh = wid >> 4;
  const float* sr = scores + (long)bh * 2 * NMODE * NMODE;
  const float* si = sr + NMODE * NMODE;
  const float* vr = ft + ((long)2 * NBH + bh) * 2 * NMODE * DHEAD;
  const float* vi = vr + NMODE * DHEAD;
  v8f cre = {}, cim = {};
  for (int k0 = 0; k0 < NMODE; k0 += 32) {
    v16h asr  = load_a_frag(sr, NMODE, tm * 16, k0, 1.0f);
    v16h asi  = load_a_frag(si, NMODE, tm * 16, k0, 1.0f);
    v16h asin = load_a_frag(si, NMODE, tm * 16, k0, -1.0f);
    v16h bvr  = load_b_frag(vr, DHEAD, 1, k0, tn * 16);
    v16h bvi  = load_b_frag(vi, DHEAD, 1, k0, tn * 16);
    cre = wmma16(asr, bvr, cre);
    cre = wmma16(asin, bvi, cre);   // sr*vr - si*vi
    cim = wmma16(asr, bvi, cim);
    cim = wmma16(asi, bvr, cim);    // sr*vi + si*vr
  }
  int lane = threadIdx.x & 31;
  int cc = lane & 15, hi = lane >> 4;
  float* X = X2 + (long)bh * 128 * DHEAD;
#pragma unroll
  for (int v = 0; v < 8; ++v) {
    int row = tm * 16 + v + 8 * hi;
    float wgt = ((row == 0) ? 1.0f : 2.0f) * (1.0f / (float)LSEQ);
    X[(long)row * DHEAD + tn * 16 + cc] = cre[v] * wgt;
    X[(long)(64 + row) * DHEAD + tn * 16 + cc] = (row == 0) ? 0.0f : cim[v] * wgt;
  }
}

// ---- K5: out[t][d] = A2[t][:] @ X2[:][d].  One block per (bh, 128-row slab):
//      A2 slab + full X2(bh) resident in LDS; only HBM stream is the output. -
#define APAD 136  // f16 row stride for K=128 (272B = 68 words; A rows land on
                  // banks 4*r -> conflict-free; 4B-aligned)
__global__ __launch_bounds__(256) void
irfft_gemm(const _Float16* __restrict__ A2, const float* __restrict__ X2,
           float* __restrict__ out) {
  __shared__ _Float16 As[128 * APAD];   // [t local][k 0..127]
  __shared__ _Float16 Xs[DHEAD * APAD]; // [d][k 0..127] (transposed, f16)
  int tid = threadIdx.x;
  int bid = blockIdx.x;                 // 0..2047
  int t0 = (bid & 31) * 128;
  int bh = bid >> 5;
  const unsigned int* Ag = (const unsigned int*)A2;   // 64 u32 per row
  const float* Xg = X2 + (long)bh * 128 * DHEAD;
  // stage A2 slab: 128*128 f16 = 8192 u32
#pragma unroll
  for (int j = 0; j < 32; ++j) {
    int c = tid + j * 256;
    int kp = c & 63, row = c >> 6;
    unsigned int d32 = Ag[(long)(t0 + row) * 64 + kp];
    *(unsigned int*)&As[row * APAD + 2 * kp] = d32;
  }
  // stage X2(bh) transposed + f32->f16: 128*64 elems = 4096 u32
#pragma unroll
  for (int j = 0; j < 16; ++j) {
    int c = tid + j * 256;
    int d = c & 63, kp = c >> 6;
    float x0 = Xg[(long)(2 * kp) * DHEAD + d];
    float x1 = Xg[(long)(2 * kp + 1) * DHEAD + d];
    *(unsigned int*)&Xs[d * APAD + 2 * kp] = pack_h2(x0, x1);
  }
  __syncthreads();
  int w = tid >> 5, lane = tid & 31;
  int r = lane & 15, hi = lane >> 4;
  v8f acc[4] = {};
#pragma unroll
  for (int k0 = 0; k0 < 128; k0 += 32) {
    v16h a = lds_a_frag(&As[(w * 16 + r) * APAD + k0], hi);
#pragma unroll
    for (int tn = 0; tn < 4; ++tn) {
      v16h b = lds_b_frag(&Xs[(tn * 16 + r) * APAD + k0], hi);
      acc[tn] = wmma16(a, b, acc[tn]);
    }
  }
  float* o = out + (long)bh * LSEQ * DHEAD;
#pragma unroll
  for (int tn = 0; tn < 4; ++tn)
    store_c_frag(o, DHEAD, t0 + w * 16, tn * 16, acc[tn]);
}

extern "C" void kernel_launch(void* const* d_in, const int* in_sizes, int n_in,
                              void* d_out, int out_size, void* d_ws, size_t ws_size,
                              hipStream_t stream) {
  const float* q = (const float*)d_in[0];
  const float* k = (const float*)d_in[1];
  const float* v = (const float*)d_in[2];
  const int* idx_q = (const int*)d_in[3];
  const int* idx_k = (const int*)d_in[4];
  const int* idx_v = (const int*)d_in[5];
  float* out = (float*)d_out;

  // workspace layout (bytes), total ~14 MB
  char* ws = (char*)d_ws;
  _Float16* Wfw = (_Float16*)(ws);            // 3*2*64*4096 f16 = 3 MB
  _Float16* A2  = (_Float16*)(ws + 3145728);  // 4096*128 f16   = 1 MB
  float* ft     = (float*)(ws + 4194304);     // 3*64*2*64*64   = 6 MB
  float* scores = (float*)(ws + 10485760);    // 64*2*64*64     = 2 MB
  float* X2     = (float*)(ws + 12582912);    // 64*128*64      = 2 MB

  build_fw_weights<<<(3 * NMODE * LSEQ + 255) / 256, 256, 0, stream>>>(
      idx_q, idx_k, idx_v, Wfw);
  build_inv_weights<<<(LSEQ * NMODE + 255) / 256, 256, 0, stream>>>(A2);

  dft_gemm<<<NBH * 3, 256, 0, stream>>>(q, k, v, Wfw, ft);   // 192 blocks
  qk_tanh<<<128, 256, 0, stream>>>(ft, scores);              // 1024 tile-waves
  attn_v<<<128, 256, 0, stream>>>(scores, ft, X2);           // 1024 tile-waves
  irfft_gemm<<<NBH * 32, 256, 0, stream>>>(A2, X2, out);     // 2048 blocks
}
